// PointNetSAModuleMSG_47571057771111
// MI455X (gfx1250) — matrix operand order repaired and checked by reading, not compiled, not run on hardware
//
#include <hip/hip_runtime.h>
#include <cstdint>
#include <cstddef>

// ---------------- problem constants (match reference) ----------------
#define B_    4
#define N_    16384
#define M_    2048
#define CIN_  64
#define EPS_  1e-5f

typedef __attribute__((ext_vector_type(16))) __bf16        v16bf;
typedef __attribute__((ext_vector_type(8)))  __bf16        v8bf;
typedef __attribute__((ext_vector_type(8)))  float         v8f;
typedef __attribute__((ext_vector_type(4)))  unsigned int  v4u;
typedef __attribute__((ext_vector_type(8)))  int           v8i;
typedef __attribute__((ext_vector_type(4)))  int           v4i;

// ---------------- small helpers (native bf16 converts) ----------------
__device__ __forceinline__ unsigned short f2bf(float f) {
  union { __bf16 b; unsigned short u; } c;
  c.b = (__bf16)f;                       // hardware RNE convert
  return c.u;
}
__device__ __forceinline__ float bf2f(unsigned short h) {
  union { unsigned short u; __bf16 b; } c;
  c.u = h;
  return (float)c.b;
}

// B-fragment swizzle: element (channel c, position p) -> flat index in Xsw.
// Per 16-wide p-tile and 32-deep k-step, lane = ((c%32)/16)*16 + (p%16) holds
// 16 contiguous bf16 (v = c%16), so each lane loads one aligned 32B fragment.
__device__ __forceinline__ size_t bswz(int c, int p, int ksteps) {
  return ((((size_t)(p >> 4) * ksteps + (c >> 5)) * 32) +
          (((c >> 4) & 1) * 16 + (p & 15))) * 16 + (c & 15);
}

// =====================================================================
// 1) Farthest point sampling: one workgroup per batch, dists in LDS.
// =====================================================================
extern __shared__ float fps_smem[];

__global__ __launch_bounds__(1024) void fps_kernel(const float* __restrict__ xyz,
                                                   int* __restrict__ idx) {
  const int b = blockIdx.x;
  float* dist = fps_smem;                       // N_ floats
  float* rv   = fps_smem + N_;                  // 1024 floats
  int*   ri   = (int*)(fps_smem + N_ + 1024);   // 1024 ints
  __shared__ float s_px, s_py, s_pz;

  const float* X = xyz + (size_t)b * 3 * N_;
  const int t = threadIdx.x;

  for (int i = t; i < N_; i += 1024) dist[i] = 1e10f;
  if (t == 0) { idx[(size_t)b * M_] = 0; s_px = X[0]; s_py = X[N_]; s_pz = X[2 * N_]; }
  __syncthreads();

  for (int it = 1; it < M_; ++it) {
    const float px = s_px, py = s_py, pz = s_pz;
    float best = -1.0f; int bi = 0x7fffffff;
    for (int i = t; i < N_; i += 1024) {
      float dx = X[i] - px, dy = X[N_ + i] - py, dz = X[2 * N_ + i] - pz;
      float d  = dx * dx + dy * dy + dz * dz;
      float dd = fminf(dist[i], d);
      dist[i] = dd;
      if (dd > best) { best = dd; bi = i; }
    }
    rv[t] = best; ri[t] = bi;
    __syncthreads();
    for (int s = 512; s > 0; s >>= 1) {
      if (t < s) {
        float o = rv[t + s]; int oi = ri[t + s];
        if (o > rv[t] || (o == rv[t] && oi < ri[t])) { rv[t] = o; ri[t] = oi; }
      }
      __syncthreads();
    }
    if (t == 0) {
      int n = ri[0];
      idx[(size_t)b * M_ + it] = n;
      s_px = X[n]; s_py = X[N_ + n]; s_pz = X[2 * N_ + n];
    }
    __syncthreads();
  }
}

__global__ __launch_bounds__(256) void gather_newxyz_kernel(const float* __restrict__ xyz,
                                                            const int* __restrict__ idx,
                                                            float* __restrict__ out) {
  int id = blockIdx.x * 256 + threadIdx.x;
  if (id >= B_ * 3 * M_) return;
  int b = id / (3 * M_);
  int r = id % (3 * M_);
  int d = r / M_, m = r % M_;
  out[id] = xyz[((size_t)b * 3 + d) * N_ + idx[(size_t)b * M_ + m]];
}

// =====================================================================
// 2) Ball query: one wave32 per query point, ballot/popc slot assignment.
// =====================================================================
__global__ __launch_bounds__(256) void ball_query_kernel(const float* __restrict__ xyz,
                                                         const float* __restrict__ nxyz,
                                                         int* __restrict__ nidx,
                                                         float r2, int K) {
  const int lane = threadIdx.x & 31;
  const int q    = blockIdx.x * 8 + (threadIdx.x >> 5);   // [0, B*M)
  const int b = q / M_, m = q % M_;
  const float* Xp = xyz + (size_t)b * 3 * N_;
  const float cx = nxyz[((size_t)b * 3 + 0) * M_ + m];
  const float cy = nxyz[((size_t)b * 3 + 1) * M_ + m];
  const float cz = nxyz[((size_t)b * 3 + 2) * M_ + m];
  int* out = nidx + (size_t)q * K;

  int cnt = 0, first = 0;
  bool havefirst = false;
  for (int base = 0; base < N_ && cnt < K; base += 32) {
    const int i = base + lane;
    float dx = Xp[i] - cx, dy = Xp[N_ + i] - cy, dz = Xp[2 * N_ + i] - cz;
    bool hit = (dx * dx + dy * dy + dz * dz) < r2;
    unsigned mask = __builtin_amdgcn_ballot_w32(hit);
    int pre = __popc(mask & ((1u << lane) - 1u));
    if (hit && (cnt + pre) < K) out[cnt + pre] = i;
    if (!havefirst && mask) { first = base + (__ffs(mask) - 1); havefirst = true; }
    cnt += __popc(mask);
  }
  if (cnt > K) cnt = K;
  if (!havefirst) first = 0;
  for (int s = cnt + lane; s < K; s += 32) out[s] = first;
}

// =====================================================================
// 3) Build grouped input directly in B-fragment-swizzled bf16 layout.
//    p = (b*M+m)*K+k, channels: 0..2 gxyz, 3..66 feature, 67..95 zero pad.
// =====================================================================
__global__ __launch_bounds__(256) void build_x_kernel(const float* __restrict__ xyz,
                                                      const float* __restrict__ feat,
                                                      const float* __restrict__ nxyz,
                                                      const int* __restrict__ nidx,
                                                      unsigned short* __restrict__ Xsw,
                                                      int K, int P) {
  size_t total = (size_t)96 * P;
  size_t id = (size_t)blockIdx.x * 256 + threadIdx.x;
  if (id >= total) return;
  int c = (int)(id % 96);
  int p = (int)(id / 96);
  int k  = p % K;
  int bm = p / K;
  int m = bm % M_, b = bm / M_;
  float v;
  if (c < 3) {
    int n = nidx[(size_t)bm * K + k];
    v = xyz[((size_t)b * 3 + c) * N_ + n] - nxyz[((size_t)b * 3 + c) * M_ + m];
  } else if (c < 3 + CIN_) {
    int n = nidx[(size_t)bm * K + k];
    v = feat[((size_t)b * CIN_ + (c - 3)) * N_ + n];
  } else {
    v = 0.0f;
  }
  Xsw[bswz(c, p, 3)] = f2bf(v);
}

// weight pack: f32 [cout, cin_real] -> bf16 in WMMA A-fragment order.
__global__ __launch_bounds__(256) void wpack_kernel(const float* __restrict__ w,
                                                    unsigned short* __restrict__ wsw,
                                                    int cout, int cin_real, int CINP) {
  int id = blockIdx.x * 256 + threadIdx.x;
  if (id >= cout * CINP) return;
  int o = id / CINP, c = id % CINP;
  int ksteps = CINP >> 5;
  int ot = o >> 4, ks = c >> 5, kl = c & 31;
  int lane = (kl >> 4) * 16 + (o & 15);
  int v = kl & 15;
  size_t di = (((size_t)(ot * ksteps + ks) * 32) + lane) * 16 + v;
  wsw[di] = (c < cin_real) ? f2bf(w[(size_t)o * cin_real + c]) : (unsigned short)0;
}

// =====================================================================
// 4) GEMM: Y[p][o] = sum_c W[o][c] * X[c][p]  via v_wmma_f32_16x16x32_bf16.
//    Weights DMA'd into LDS by the Tensor Data Mover; fragments are single
//    aligned 32B loads; output converted with native packed cvt and written
//    as one b128 store per tile.
// =====================================================================
template <int NO, int KSTEPS>
__global__ __launch_bounds__(256) void gemm_bf16_kernel(const unsigned short* __restrict__ Xsw,
                                                        const unsigned short* __restrict__ Wsw,
                                                        unsigned short* __restrict__ Y,
                                                        int P) {
  constexpr int COUT = NO * 16;
  constexpr int WTOT = COUT * (KSTEPS * 32);
  __shared__ __attribute__((aligned(32))) unsigned short wlds[WTOT];

#if __has_builtin(__builtin_amdgcn_tensor_load_to_lds) && __has_builtin(__builtin_amdgcn_s_wait_tensorcnt)
  if (threadIdx.x < 32) {
    unsigned long long ga = (unsigned long long)(size_t)Wsw;
    unsigned int lo = (unsigned int)(size_t)&wlds[0];
    // D# group 0: count=1, lds_addr, 57-bit global_addr, type=2 ("image")
    v4u g0 = { 1u, lo, (unsigned int)ga,
               (unsigned int)((ga >> 32) & 0x1FFFFFFu) | (2u << 30) };
    // D# group 1: data_size=2B; 1-D tile of WTOT elements (tile_dim1=0)
    v8i g1 = { (int)0x00010000,
               (int)((WTOT & 0xFFFF) << 16),                 // tensor_dim0 lo
               (int)(((WTOT >> 16) & 0xFFFF) | (1 << 16)),   // dim0 hi | tensor_dim1=1
               (int)((WTOT & 0xFFFF) << 16),                 // tile_dim0 = WTOT
               0,                                            // tile_dim1=0 (1-D)
               (int)WTOT,                                    // tensor_dim0_stride
               0, 0 };
    v4i g2 = { 0, 0, 0, 0 };
    v4i g3 = { 0, 0, 0, 0 };
#if __clang_major__ >= 23
    v8i g4 = { 0, 0, 0, 0, 0, 0, 0, 0 };
    __builtin_amdgcn_tensor_load_to_lds(g0, g1, g2, g3, g4, 0);
#else
    __builtin_amdgcn_tensor_load_to_lds(g0, g1, g2, g3, 0);
#endif
    __builtin_amdgcn_s_wait_tensorcnt(0);
  }
#else
  for (int i = threadIdx.x; i < WTOT; i += 256) wlds[i] = Wsw[i];
#endif
  __syncthreads();

  const int lane = threadIdx.x & 31;
  const int wave = threadIdx.x >> 5;
  const int pt = blockIdx.x * 8 + wave;
  const int p  = pt * 16 + (lane & 15);
  const int kh = lane >> 4;

  const unsigned short* xb = Xsw + ((size_t)pt * KSTEPS * 32 + lane) * 16;

  v8f acc[NO] = {};
#pragma unroll
  for (int ks = 0; ks < KSTEPS; ++ks) {
    if (ks + 1 < KSTEPS) __builtin_prefetch(xb + (size_t)(ks + 1) * 512, 0, 1);
    v16bf bv = *(const v16bf*)(xb + (size_t)ks * 512);     // 32B coalesced
#pragma unroll
    for (int ot = 0; ot < NO; ++ot) {
      v16bf av = *(const v16bf*)(wlds + ((ot * KSTEPS + ks) * 32 + lane) * 16);
      acc[ot] = __builtin_amdgcn_wmma_f32_16x16x32_bf16(
          false, av, false, bv, (short)0, acc[ot], false, false);
    }
  }

  // C/D layout: VGPR r holds o = ot*16 + kh*8 + r  ->  8 consecutive channels.
  // Native packed f32->bf16 convert, single 16B store per tile.
#pragma unroll
  for (int ot = 0; ot < NO; ++ot) {
    v8bf d;
#pragma unroll
    for (int r = 0; r < 8; ++r) d[r] = (__bf16)acc[ot][r];
    *(v8bf*)(Y + (size_t)p * COUT + ot * 16 + kh * 8) = d;
  }
}

// =====================================================================
// 5) BatchNorm stats / finalize / apply / final max-over-k (Y is p-major)
//    stats: [0..127]=sum [128..255]=sumsq [256..383]=a [384..511]=bshift
// =====================================================================
__global__ __launch_bounds__(512) void zero_stats_kernel(float* stats) {
  if (threadIdx.x < 512) stats[threadIdx.x] = 0.0f;
}

// blockDim = 2*cout (128/192/256): thread t -> channel t%cout, row parity t/cout
__global__ __launch_bounds__(256) void bn_stats_kernel(const unsigned short* __restrict__ Y,
                                                       float* __restrict__ stats,
                                                       int P, int cout) {
  int t = threadIdx.x;
  int c = t % cout, pr = t / cout;
  float s = 0.f, q = 0.f;
  for (int p = blockIdx.x * 2 + pr; p < P; p += gridDim.x * 2) {
    float v = bf2f(Y[(size_t)p * cout + c]);
    s += v; q += v * v;
  }
  __shared__ float ssum[256], ssq[256];
  ssum[t] = s; ssq[t] = q;
  __syncthreads();
  if (pr == 0) {
    atomicAdd(&stats[c],       ssum[c] + ssum[c + cout]);
    atomicAdd(&stats[128 + c], ssq[c]  + ssq[c + cout]);
  }
}

__global__ __launch_bounds__(128) void bn_finalize_kernel(float* __restrict__ stats,
                                                          const float* __restrict__ g,
                                                          const float* __restrict__ bb,
                                                          int P, int cout) {
  int c = threadIdx.x;
  if (c >= cout) return;
  float inv = 1.0f / (float)P;
  float mu  = stats[c] * inv;
  float var = stats[128 + c] * inv - mu * mu;
  float a   = g[c] * rsqrtf(var + EPS_);
  stats[256 + c] = a;
  stats[384 + c] = bb[c] - mu * a;
}

// normalize+relu, re-cast to bf16 into the NEXT layer's swizzled input
__global__ __launch_bounds__(256) void bn_apply_kernel(const unsigned short* __restrict__ Y,
                                                       const float* __restrict__ stats,
                                                       unsigned short* __restrict__ Xsw,
                                                       int P, int cout, int ksteps_next) {
  size_t total = (size_t)cout * P;
  size_t id = (size_t)blockIdx.x * 256 + threadIdx.x;
  if (id >= total) return;
  int c = (int)(id % cout);
  int p = (int)(id / cout);
  float v = stats[256 + c] * bf2f(Y[id]) + stats[384 + c];
  Xsw[bswz(c, p, ksteps_next)] = f2bf(v > 0.f ? v : 0.f);
}

// wave per (b,m): lanes own channels {lane, lane+32, lane+64, lane+96}; rows
// of Y are contiguous (K*128 bf16), so reads are fully coalesced.
__global__ __launch_bounds__(256) void bn_max_kernel(const unsigned short* __restrict__ Y,
                                                     const float* __restrict__ stats,
                                                     float* __restrict__ out,
                                                     int K, int coff) {
  const int lane = threadIdx.x & 31;
  const int q = blockIdx.x * 8 + (threadIdx.x >> 5);   // (b,m) in [0, B*M)
  const int b = q / M_, m = q % M_;
  float a[4], bs[4], mx[4];
#pragma unroll
  for (int j = 0; j < 4; ++j) {
    a[j]  = stats[256 + lane + 32 * j];
    bs[j] = stats[384 + lane + 32 * j];
    mx[j] = 0.0f;                                // relu >= 0 identity
  }
  for (int k = 0; k < K; ++k) {
    const unsigned short* row = Y + ((size_t)q * K + k) * 128;
#pragma unroll
    for (int j = 0; j < 4; ++j) {
      float v = a[j] * bf2f(row[lane + 32 * j]) + bs[j];
      v = v > 0.f ? v : 0.f;
      mx[j] = fmaxf(mx[j], v);
    }
  }
#pragma unroll
  for (int j = 0; j < 4; ++j)
    out[(size_t)(B_ * 3 * M_) + ((size_t)b * 256 + coff + lane + 32 * j) * M_ + m] = mx[j];
}

// =====================================================================
// Host orchestration
// =====================================================================
static inline size_t align256(size_t x) { return (x + 255) & ~(size_t)255; }

extern "C" void kernel_launch(void* const* d_in, const int* in_sizes, int n_in,
                              void* d_out, int out_size, void* d_ws, size_t ws_size,
                              hipStream_t stream) {
  (void)in_sizes; (void)n_in; (void)out_size; (void)ws_size;
  const float* xyz  = (const float*)d_in[0];
  const float* feat = (const float*)d_in[1];
  float* out = (float*)d_out;

  auto Wp = [&](int s, int l) { return (const float*)d_in[2 + (s * 3 + l) * 3 + 0]; };
  auto Gp = [&](int s, int l) { return (const float*)d_in[2 + (s * 3 + l) * 3 + 1]; };
  auto Bp = [&](int s, int l) { return (const float*)d_in[2 + (s * 3 + l) * 3 + 2]; };

  char* ws = (char*)d_ws;
  size_t off = 0;
  int* idxbuf = (int*)(ws + off);          off = align256(off + (size_t)B_ * M_ * 4);
  int* nidx   = (int*)(ws + off);          off = align256(off + (size_t)B_ * M_ * 64 * 4);
  float* stats = (float*)(ws + off);       off = align256(off + 512 * 4);
  unsigned short* wsw = (unsigned short*)(ws + off); off = align256(off + 128 * 96 * 2);
  unsigned short* Xsw = (unsigned short*)(ws + off); off = align256(off + (size_t)96 * B_ * M_ * 64 * 2);
  unsigned short* Yb  = (unsigned short*)(ws + off); off = align256(off + (size_t)128 * B_ * M_ * 64 * 2);

  // ---- FPS + new_xyz ----
  size_t fps_lds = (size_t)N_ * 4 + 1024 * 4 + 1024 * 4;
  fps_kernel<<<B_, 1024, fps_lds, stream>>>(xyz, idxbuf);
  gather_newxyz_kernel<<<(B_ * 3 * M_ + 255) / 256, 256, 0, stream>>>(xyz, idxbuf, out);

  const int   KS_[2]     = {32, 64};
  const float RAD_[2]    = {0.2f, 0.4f};
  const int   MLP_[2][3] = {{64, 64, 128}, {64, 96, 128}};

  for (int s = 0; s < 2; ++s) {
    const int K = KS_[s];
    const int P = B_ * M_ * K;
    const float r2 = RAD_[s] * RAD_[s];

    ball_query_kernel<<<(B_ * M_) / 8, 256, 0, stream>>>(xyz, out, nidx, r2, K);

    {
      size_t total = (size_t)96 * P;
      build_x_kernel<<<(unsigned)((total + 255) / 256), 256, 0, stream>>>(
          xyz, feat, out, nidx, Xsw, K, P);
    }

    int cin_real = 67, CINP = 96;
    for (int l = 0; l < 3; ++l) {
      const int cout = MLP_[s][l];
      wpack_kernel<<<(cout * CINP + 255) / 256, 256, 0, stream>>>(Wp(s, l), wsw, cout, cin_real, CINP);
      zero_stats_kernel<<<1, 512, 0, stream>>>(stats);

      const int gblocks = P / 128;    // 8 waves/block x 16 positions/wave
      if (CINP == 96 && cout == 64)        gemm_bf16_kernel<4, 3><<<gblocks, 256, 0, stream>>>(Xsw, wsw, Yb, P);
      else if (CINP == 64 && cout == 64)   gemm_bf16_kernel<4, 2><<<gblocks, 256, 0, stream>>>(Xsw, wsw, Yb, P);
      else if (CINP == 64 && cout == 96)   gemm_bf16_kernel<6, 2><<<gblocks, 256, 0, stream>>>(Xsw, wsw, Yb, P);
      else if (CINP == 64 && cout == 128)  gemm_bf16_kernel<8, 2><<<gblocks, 256, 0, stream>>>(Xsw, wsw, Yb, P);
      else /* CINP==96, cout==128 */       gemm_bf16_kernel<8, 3><<<gblocks, 256, 0, stream>>>(Xsw, wsw, Yb, P);

      bn_stats_kernel<<<512, 2 * cout, 0, stream>>>(Yb, stats, P, cout);
      bn_finalize_kernel<<<1, 128, 0, stream>>>(stats, Gp(s, l), Bp(s, l), P, cout);

      if (l < 2) {
        size_t total = (size_t)cout * P;
        bn_apply_kernel<<<(unsigned)((total + 255) / 256), 256, 0, stream>>>(
            Yb, stats, Xsw, P, cout, MLP_[s][l] / 32);
      } else {
        bn_max_kernel<<<(B_ * M_) / 8, 256, 0, stream>>>(Yb, stats, out, K, s * 128);
      }
      cin_real = cout;
      CINP = cout;
    }
  }
}
